// HGTStyleDetector_47888885350870
// MI455X (gfx1250) — matrix operand order repaired and checked by reading, not compiled, not run on hardware
//
#include <hip/hip_runtime.h>
#include <hip/hip_bf16.h>

#define DIN   128
#define HID   256
#define HEADS 8
#define CCH   32
#define GG    16

typedef __attribute__((ext_vector_type(16))) __bf16 v16bf;
typedef __attribute__((ext_vector_type(8)))  float  v8f;

union BfVec { v16bf bf; uint4 q[2]; };

__device__ __forceinline__ unsigned short f2bf(float f) {
  unsigned u = __float_as_uint(f);
  u += 0x7FFFu + ((u >> 16) & 1u);          // round-to-nearest-even bf16
  return (unsigned short)(u >> 16);
}
__device__ __forceinline__ unsigned f2ord(float f) {   // monotone float->u32
  unsigned u = __float_as_uint(f);
  return (u & 0x80000000u) ? ~u : (u | 0x80000000u);
}
__device__ __forceinline__ float ord2f(unsigned u) {
  return (u & 0x80000000u) ? __uint_as_float(u & 0x7FFFFFFFu) : __uint_as_float(~u);
}

// ---------------- conversions ----------------
__global__ void k_f32_to_bf16(const float* __restrict__ in,
                              unsigned short* __restrict__ out, int n) {
  int i = blockIdx.x * blockDim.x + threadIdx.x;
  if (i < n) out[i] = f2bf(in[i]);
}

// W[K][Nc] f32  ->  Wt[Nc][K] bf16 (transpose so GEMM B loads are contiguous)
__global__ void k_wt_bf16(const float* __restrict__ w,
                          unsigned short* __restrict__ wt, int K, int Nc) {
  int i = blockIdx.x * blockDim.x + threadIdx.x;
  if (i >= K * Nc) return;
  int k = i / Nc, n = i - k * Nc;
  wt[(size_t)n * K + k] = f2bf(w[i]);
}

// ---------------- WMMA GEMM: C[M][Nc] = act(A_bf16[M][K] @ Bt_bf16[Nc][K]^T + bias) --------
// One wave owns a 16x64 C footprint: 4 accumulators, A fragment reused across 4 WMMAs.
// All 5 fragments (A + 4xB) are loaded into distinct registers before the 4 WMMAs so the
// load clause overlaps the WMMA pipe (staged s_wait_loadcnt instead of full drains).
// A fragment layout (16-bit A 16x32): lane<16 -> K[0..15], lane>=16 -> K[16..31]
// B fragment layout (16-bit B 32x16): lane<16 -> K[0..15], lane>=16 -> K[16..31]
__global__ __launch_bounds__(256) void k_wmma_gemm(
    const unsigned short* __restrict__ A,   // [M][K] bf16
    const unsigned short* __restrict__ Bt,  // [Nc][K] bf16
    const float* __restrict__ bias,         // [Nc]
    float* __restrict__ C,                  // [M][Nc] f32
    int M, int K, int Nc, int relu) {
  int tile = (blockIdx.x * blockDim.x + threadIdx.x) >> 5;
  int lane = threadIdx.x & 31;
  int tilesNg = Nc >> 6;                      // groups of 64 columns (4 x 16)
  int totalTiles = (M >> 4) * tilesNg;
  if (tile >= totalTiles) return;
  int tm = tile / tilesNg, tg = tile - tm * tilesNg;
  int half = lane >> 4;
  int l15  = lane & 15;

  const unsigned short* arow = A + (size_t)(tm * 16 + l15) * K + half * 16;
  const unsigned short* b0p = Bt + (size_t)(tg * 64 +  0 + l15) * K + half * 16;
  const unsigned short* b1p = Bt + (size_t)(tg * 64 + 16 + l15) * K + half * 16;
  const unsigned short* b2p = Bt + (size_t)(tg * 64 + 32 + l15) * K + half * 16;
  const unsigned short* b3p = Bt + (size_t)(tg * 64 + 48 + l15) * K + half * 16;

  v8f acc0 = {}, acc1 = {}, acc2 = {}, acc3 = {};
  for (int k0 = 0; k0 < K; k0 += 32) {
    __builtin_prefetch(arow + k0 + 32, 0, 0);
    BfVec a, b0, b1, b2, b3;
    // issue the full load clause first, into distinct registers
    a.q[0]  = *(const uint4*)(arow + k0);        // K = k0 + half*16 + [0..7]
    a.q[1]  = *(const uint4*)(arow + k0 + 8);    // K = k0 + half*16 + 8 + [0..7]
    b0.q[0] = *(const uint4*)(b0p + k0);
    b0.q[1] = *(const uint4*)(b0p + k0 + 8);
    b1.q[0] = *(const uint4*)(b1p + k0);
    b1.q[1] = *(const uint4*)(b1p + k0 + 8);
    b2.q[0] = *(const uint4*)(b2p + k0);
    b2.q[1] = *(const uint4*)(b2p + k0 + 8);
    b3.q[0] = *(const uint4*)(b3p + k0);
    b3.q[1] = *(const uint4*)(b3p + k0 + 8);
    acc0 = __builtin_amdgcn_wmma_f32_16x16x32_bf16(false, a.bf, false, b0.bf, (short)0, acc0, false, false);
    acc1 = __builtin_amdgcn_wmma_f32_16x16x32_bf16(false, a.bf, false, b1.bf, (short)0, acc1, false, false);
    acc2 = __builtin_amdgcn_wmma_f32_16x16x32_bf16(false, a.bf, false, b2.bf, (short)0, acc2, false, false);
    acc3 = __builtin_amdgcn_wmma_f32_16x16x32_bf16(false, a.bf, false, b3.bf, (short)0, acc3, false, false);
  }

  int m0 = tm * 16 + half * 8;
  v8f accs[4] = {acc0, acc1, acc2, acc3};
#pragma unroll
  for (int j = 0; j < 4; ++j) {
    int n = tg * 64 + j * 16 + l15;
    float bv = bias[n];
#pragma unroll
    for (int r = 0; r < 8; ++r) {
      float v = accs[j][r] + bv;
      if (relu) v = v > 0.f ? v : 0.f;
      C[(size_t)(m0 + r) * Nc + n] = v;
    }
  }
}

// ---------------- GATv2 edge phase ----------------
// score[e][h] = sum_c leakyrelu(xl[src]+xr[dst], 0.2) * att[h][c] ; segment max via u32 atomics
__global__ void k_edge_score(const int* __restrict__ ei, int E, int ET,
                             const float* __restrict__ xl, const float* __restrict__ xr,
                             const float* __restrict__ att,
                             float* __restrict__ score, unsigned* __restrict__ smaxu) {
  int idx = blockIdx.x * blockDim.x + threadIdx.x;
  if (idx >= ET * HEADS) return;
  int e = idx >> 3, hd = idx & 7;
  int s = (e < E) ? ei[e]     : (e - E);
  int d = (e < E) ? ei[E + e] : (e - E);
  const float* pl = xl + (size_t)s * HID + hd * CCH;
  const float* pr = xr + (size_t)d * HID + hd * CCH;
  const float* pa = att + hd * CCH;
  float sum = 0.f;
#pragma unroll
  for (int c = 0; c < CCH; ++c) {
    float v = pl[c] + pr[c];
    v = (v > 0.f) ? v : 0.2f * v;
    sum += v * pa[c];
  }
  score[idx] = sum;
  atomicMax(smaxu + (size_t)d * HEADS + hd, f2ord(sum));   // global_atomic_max_u32
}

__global__ void k_edge_exp(const int* __restrict__ ei, int E, int ET,
                           float* __restrict__ score, const unsigned* __restrict__ smaxu,
                           float* __restrict__ denom) {
  int idx = blockIdx.x * blockDim.x + threadIdx.x;
  if (idx >= ET * HEADS) return;
  int e = idx >> 3, hd = idx & 7;
  int d = (e < E) ? ei[E + e] : (e - E);
  float sm = ord2f(smaxu[(size_t)d * HEADS + hd]);
  float ex = __expf(score[idx] - sm);
  score[idx] = ex;                                          // overwrite score with exp
  atomicAdd(denom + (size_t)d * HEADS + hd, ex);            // global_atomic_add_f32
}

// gout[dst][ch] += (ex/denom[dst][h]) * xl[src][ch]; one block per edge, 256 ch threads
__global__ __launch_bounds__(256) void k_edge_aggregate(
    const int* __restrict__ ei, int E,
    const float* __restrict__ ex, const float* __restrict__ denom,
    const float* __restrict__ xl, float* __restrict__ gout) {
  int e  = blockIdx.x;
  int ch = threadIdx.x;
  int s = (e < E) ? ei[e]     : (e - E);
  int d = (e < E) ? ei[E + e] : (e - E);
  int hd = ch >> 5;
  float alpha = ex[(size_t)e * HEADS + hd] / denom[(size_t)d * HEADS + hd];
  atomicAdd(gout + (size_t)d * HID + ch, alpha * xl[(size_t)s * HID + ch]);
}

__global__ void k_bias_relu(const float* __restrict__ in, const float* __restrict__ bias,
                            float* __restrict__ out, int total) {
  int i = blockIdx.x * blockDim.x + threadIdx.x;
  if (i >= total) return;
  float v = in[i] + bias[i & (HID - 1)];
  out[i] = v > 0.f ? v : 0.f;
}

// ---------------- LayerNorm: one wave32 per node (HID=256 -> 8 vals/lane) ----------------
__global__ __launch_bounds__(256) void k_layernorm(
    const float* __restrict__ in, const float* __restrict__ g, const float* __restrict__ b,
    float* __restrict__ out, unsigned short* __restrict__ outbf, int n) {
  int wave = (blockIdx.x * blockDim.x + threadIdx.x) >> 5;
  int lane = threadIdx.x & 31;
  if (wave >= n) return;
  const float* row = in + (size_t)wave * HID;
  float v[8];
  float s = 0.f;
#pragma unroll
  for (int i = 0; i < 8; ++i) { v[i] = row[lane + i * 32]; s += v[i]; }
#pragma unroll
  for (int off = 16; off > 0; off >>= 1) s += __shfl_xor(s, off, 32);
  float mu = s * (1.0f / HID);
  float var = 0.f;
#pragma unroll
  for (int i = 0; i < 8; ++i) { float d = v[i] - mu; var += d * d; }
#pragma unroll
  for (int off = 16; off > 0; off >>= 1) var += __shfl_xor(var, off, 32);
  float rs = rsqrtf(var * (1.0f / HID) + 1e-5f);
#pragma unroll
  for (int i = 0; i < 8; ++i) {
    int ch = lane + i * 32;
    float o = (v[i] - mu) * rs * g[ch] + b[ch];
    out[(size_t)wave * HID + ch] = o;
    outbf[(size_t)wave * HID + ch] = f2bf(o);
  }
}

// ---------------- pooling + head ----------------
__global__ void k_pool_sum(const float* __restrict__ h, const int* __restrict__ batch,
                           float* __restrict__ pooled, float* __restrict__ cnt, int n) {
  int i = blockIdx.x * blockDim.x + threadIdx.x;
  if (i >= n * HID) return;
  int node = i >> 8, ch = i & 255;
  int gi = batch[node];
  atomicAdd(pooled + (size_t)gi * HID + ch, h[i]);
  if (ch == 0) atomicAdd(cnt + gi, 1.0f);
}

__global__ void k_pool_div(float* __restrict__ pooled, const float* __restrict__ cnt,
                           unsigned short* __restrict__ pooledbf) {
  int i = blockIdx.x * blockDim.x + threadIdx.x;
  if (i >= GG * HID) return;
  float c = cnt[i >> 8]; c = c > 1.f ? c : 1.f;
  float v = pooled[i] / c;
  pooledbf[i] = f2bf(v);
}

__global__ void k_head2(const float* __restrict__ mid, const float* __restrict__ w,
                        const float* __restrict__ b, float* __restrict__ out) {
  int i = threadIdx.x;
  if (i >= GG * 2) return;
  int g = i >> 1, o = i & 1;
  float s = b[o];
#pragma unroll 8
  for (int k = 0; k < 128; ++k) s += mid[g * 128 + k] * w[k * 2 + o];
  out[i] = s;
}

// ---------------- host launch ----------------
static inline size_t al256(size_t x) { return (x + 255) & ~(size_t)255; }

extern "C" void kernel_launch(void* const* d_in, const int* in_sizes, int n_in,
                              void* d_out, int out_size, void* d_ws, size_t ws_size,
                              hipStream_t stream) {
  const float* x       = (const float*)d_in[0];
  const int*   ei      = (const int*)d_in[1];
  const int*   batch   = (const int*)d_in[2];
  const float* enc_w   = (const float*)d_in[3];
  const float* enc_b   = (const float*)d_in[4];
  const float* g1_wl   = (const float*)d_in[5];
  const float* g1_bl   = (const float*)d_in[6];
  const float* g1_wr   = (const float*)d_in[7];
  const float* g1_br   = (const float*)d_in[8];
  const float* g1_att  = (const float*)d_in[9];
  const float* g1_bias = (const float*)d_in[10];
  const float* g2_wl   = (const float*)d_in[11];
  const float* g2_bl   = (const float*)d_in[12];
  const float* g2_wr   = (const float*)d_in[13];
  const float* g2_br   = (const float*)d_in[14];
  const float* g2_att  = (const float*)d_in[15];
  const float* g2_bias = (const float*)d_in[16];
  const float* ln_g    = (const float*)d_in[17];
  const float* ln_b    = (const float*)d_in[18];
  const float* h1_w    = (const float*)d_in[19];
  const float* h1_b    = (const float*)d_in[20];
  const float* h2_w    = (const float*)d_in[21];
  const float* h2_b    = (const float*)d_in[22];

  const int N  = in_sizes[0] / DIN;
  const int E  = in_sizes[1] / 2;
  const int ET = E + N;                  // + self loops

  // workspace carve
  char* base = (char*)d_ws; size_t off = 0;
  auto alloc = [&](size_t bytes) { void* r = base + off; off = al256(off + bytes); return r; };
  unsigned short* xbf   = (unsigned short*)alloc((size_t)N * DIN * 2);
  unsigned short* hbf   = (unsigned short*)alloc((size_t)N * HID * 2);
  float* h     = (float*)alloc((size_t)N * HID * 4);
  float* xl    = (float*)alloc((size_t)N * HID * 4);
  float* xr    = (float*)alloc((size_t)N * HID * 4);
  float* gout  = (float*)alloc((size_t)N * HID * 4);
  float* score = (float*)alloc((size_t)ET * HEADS * 4);
  unsigned* smaxu = (unsigned*)alloc((size_t)N * HEADS * 4);
  float* denom = (float*)alloc((size_t)N * HEADS * 4);
  unsigned short* wencT = (unsigned short*)alloc((size_t)HID * DIN * 2);
  unsigned short* w1lT  = (unsigned short*)alloc((size_t)HID * HID * 2);
  unsigned short* w1rT  = (unsigned short*)alloc((size_t)HID * HID * 2);
  unsigned short* w2lT  = (unsigned short*)alloc((size_t)HID * HID * 2);
  unsigned short* w2rT  = (unsigned short*)alloc((size_t)HID * HID * 2);
  unsigned short* wh1T  = (unsigned short*)alloc((size_t)128 * HID * 2);
  float* pooled = (float*)alloc((size_t)GG * HID * 4);
  float* cnt    = (float*)alloc((size_t)GG * 4);
  unsigned short* pooledbf = (unsigned short*)alloc((size_t)GG * HID * 2);
  float* mid    = (float*)alloc((size_t)GG * 128 * 4);

  const int T = 256;
  auto blk = [](long long n, int t) { return (int)((n + t - 1) / t); };

  // weight / input conversion to bf16 (transposed weights for contiguous B loads)
  k_wt_bf16<<<blk(DIN * HID, T), T, 0, stream>>>(enc_w, wencT, DIN, HID);
  k_wt_bf16<<<blk(HID * HID, T), T, 0, stream>>>(g1_wl, w1lT, HID, HID);
  k_wt_bf16<<<blk(HID * HID, T), T, 0, stream>>>(g1_wr, w1rT, HID, HID);
  k_wt_bf16<<<blk(HID * HID, T), T, 0, stream>>>(g2_wl, w2lT, HID, HID);
  k_wt_bf16<<<blk(HID * HID, T), T, 0, stream>>>(g2_wr, w2rT, HID, HID);
  k_wt_bf16<<<blk(HID * 128, T), T, 0, stream>>>(h1_w, wh1T, HID, 128);
  k_f32_to_bf16<<<blk((long long)N * DIN, T), T, 0, stream>>>(x, xbf, N * DIN);

  // encoder: h = relu(x @ enc_w + enc_b)
  {
    int tiles = (N / 16) * (HID / 64);
    k_wmma_gemm<<<blk((long long)tiles * 32, T), T, 0, stream>>>(
        xbf, wencT, enc_b, h, N, DIN, HID, 1);
  }
  k_f32_to_bf16<<<blk((long long)N * HID, T), T, 0, stream>>>(h, hbf, N * HID);

  auto gat_layer = [&](const unsigned short* wlT, const float* bl,
                       const unsigned short* wrT, const float* br,
                       const float* att, const float* bias) {
    int tiles = (N / 16) * (HID / 64);
    k_wmma_gemm<<<blk((long long)tiles * 32, T), T, 0, stream>>>(hbf, wlT, bl, xl, N, HID, HID, 0);
    k_wmma_gemm<<<blk((long long)tiles * 32, T), T, 0, stream>>>(hbf, wrT, br, xr, N, HID, HID, 0);
    hipMemsetAsync(smaxu, 0, (size_t)N * HEADS * 4, stream);
    hipMemsetAsync(denom, 0, (size_t)N * HEADS * 4, stream);
    hipMemsetAsync(gout,  0, (size_t)N * HID * 4, stream);
    k_edge_score<<<blk((long long)ET * HEADS, T), T, 0, stream>>>(ei, E, ET, xl, xr, att, score, smaxu);
    k_edge_exp<<<blk((long long)ET * HEADS, T), T, 0, stream>>>(ei, E, ET, score, smaxu, denom);
    k_edge_aggregate<<<ET, 256, 0, stream>>>(ei, E, score, denom, xl, gout);
    k_bias_relu<<<blk((long long)N * HID, T), T, 0, stream>>>(gout, bias, h, N * HID);
  };

  // GAT layer 1 -> layernorm (writes h + hbf) -> GAT layer 2
  gat_layer(w1lT, g1_bl, w1rT, g1_br, g1_att, g1_bias);
  k_layernorm<<<blk((long long)N * 32, T), T, 0, stream>>>(h, ln_g, ln_b, h, hbf, N);
  gat_layer(w2lT, g2_bl, w2rT, g2_br, g2_att, g2_bias);

  // global mean pool
  hipMemsetAsync(pooled, 0, (size_t)GG * HID * 4, stream);
  hipMemsetAsync(cnt,    0, (size_t)GG * 4, stream);
  k_pool_sum<<<blk((long long)N * HID, T), T, 0, stream>>>(h, batch, pooled, cnt, N);
  k_pool_div<<<blk(GG * HID, T), T, 0, stream>>>(pooled, cnt, pooledbf);

  // head: mid = relu(pooled @ h1_w + h1_b)  (16x256x128 via WMMA: 2 waves), then 16x128x2
  k_wmma_gemm<<<1, 256, 0, stream>>>(pooledbf, wh1T, h1_b, mid, GG, HID, 128, 1);
  k_head2<<<1, 32, 0, stream>>>(mid, h2_w, h2_b, (float*)d_out);
}
